// SABlock_47055661695314
// MI455X (gfx1250) — matrix operand order repaired and checked by reading, not compiled, not run on hardware
//
#include <hip/hip_runtime.h>
#include <hip/hip_bf16.h>
#include <math.h>

typedef __attribute__((ext_vector_type(16))) __bf16 bf16x16;
typedef __attribute__((ext_vector_type(8)))  __bf16 bf16x8;
typedef __attribute__((ext_vector_type(8)))  float  f32x8;
typedef int v4i __attribute__((vector_size(16)));   // matches builtin's V4i

#ifndef __has_builtin
#define __has_builtin(x) 0
#endif

#if defined(__gfx1250__) && __has_builtin(__builtin_amdgcn_global_load_async_to_lds_b128)
#define USE_ASYNC_LDS 1
#else
#define USE_ASYNC_LDS 0
#endif

namespace {
constexpr int   kHidden = 1024;
constexpr int   kHeads  = 16;
constexpr int   kHdim   = 64;
constexpr int   kSeq    = 2048;
constexpr int   kBatch  = 2;
constexpr int   kMtot   = kBatch * kSeq;     // 4096 rows
constexpr float kScale  = 0.125f;            // 1/sqrt(64), exact in bf16
}

// ---------------------------------------------------------------------------
// WMMA helper: D = A(16x32 bf16) * B(32x16 bf16) + C(16x16 f32)
// ---------------------------------------------------------------------------
__device__ __forceinline__ f32x8 wmma_bf16f32(bf16x16 a, bf16x16 b, f32x8 c) {
  return __builtin_amdgcn_wmma_f32_16x16x32_bf16(
      /*neg_a=*/false, a, /*neg_b=*/false, b,
      /*c_mod=*/(short)0, c, /*reuse_a=*/false, /*reuse_b=*/false);
}

// Build a 16-element bf16 fragment from two contiguous 8-element runs.
__device__ __forceinline__ bf16x16 load2x8(const __bf16* p0, const __bf16* p1) {
  bf16x8 r0 = *(const bf16x8*)p0;
  bf16x8 r1 = *(const bf16x8*)p1;
  bf16x16 r;
#pragma unroll
  for (int j = 0; j < 8; ++j) { r[j] = r0[j]; r[8 + j] = r1[j]; }
  return r;
}

// Same, but converting from fp32 source on the fly (lowered to v_cvt_pk_bf16).
__device__ __forceinline__ bf16x16 cvt2x8(const float* p0, const float* p1) {
  bf16x16 r;
#pragma unroll
  for (int j = 0; j < 8; ++j) { r[j] = (__bf16)p0[j]; r[8 + j] = (__bf16)p1[j]; }
  return r;
}

// 16-byte global -> LDS copy: async (ASYNCcnt-tracked) on CDNA5 toolchains
// that expose the builtin, otherwise a synchronous b128 load/store pair.
__device__ __forceinline__ void cp16_g2s(float* dst_lds, const float* src_g) {
#if USE_ASYNC_LDS
  __builtin_amdgcn_global_load_async_to_lds_b128(
      (__attribute__((address_space(1))) v4i*)src_g,
      (__attribute__((address_space(3))) v4i*)dst_lds, 0, 0);
#else
  *(float4*)dst_lds = *(const float4*)src_g;
#endif
}

template <int N>
__device__ __forceinline__ void wait_async() {
#if USE_ASYNC_LDS
#if __has_builtin(__builtin_amdgcn_s_wait_asynccnt)
  __builtin_amdgcn_s_wait_asynccnt(N);
#else
  asm volatile("s_wait_asynccnt %0" ::"i"(N) : "memory");
#endif
#endif
}

// Cross-lane reductions within each 16-lane half of a wave32
// (C-matrix rows are striped across lanes 0-15 / 16-31).
__device__ __forceinline__ float half_max(float v) {
#pragma unroll
  for (int m = 1; m < 16; m <<= 1) v = fmaxf(v, __shfl_xor(v, m, 32));
  return v;
}
__device__ __forceinline__ float half_sum(float v) {
#pragma unroll
  for (int m = 1; m < 16; m <<= 1) v += __shfl_xor(v, m, 32);
  return v;
}

// ---------------------------------------------------------------------------
// Kernel 1: fused QKV projection, LDS-staged GEMM.
//   y = x @ W^T  (W [N,K] row-major -> B column n is row n of W).
// Block = 256 threads (8 waves, 2x4) computing a 64x128 output tile.
// Each 32-deep k stage: A(64x32 f32, 8KB) + B(128x32 f32, 16KB) cooperatively
// copied global->LDS (async-to-LDS when available, double-buffered with
// s_wait_asynccnt), then each wave runs 4 WMMA on a 32x32 subtile.
// Q stored pre-scaled bf16 [bh][t][d]; K bf16 [bh][t][d]; V bf16 [bh][d][t].
// ---------------------------------------------------------------------------
__global__ __launch_bounds__(256) void qkv_proj_kernel(
    const float* __restrict__ X, const float* __restrict__ Ctx,
    const float* __restrict__ Wq, const float* __restrict__ Wk,
    const float* __restrict__ Wv,
    __bf16* __restrict__ q_ws, __bf16* __restrict__ k_ws,
    __bf16* __restrict__ vT_ws) {
  __shared__ __align__(16) float lds_a[2][64][32];   // 2 x 8KB
  __shared__ __align__(16) float lds_b[2][128][32];  // 2 x 16KB

  const int tid  = threadIdx.x;
  const int wave = tid >> 5;
  const int lane = tid & 31;
  const int l = lane & 15, hi = lane >> 4;
  const int wm = wave >> 2, wn = wave & 3;
  const int mblk = blockIdx.x * 64;
  const int nblk = blockIdx.y * 128;
  const int which = blockIdx.z;  // 0=Q (from x), 1=K, 2=V (from context)

  const float* A = (which == 0) ? X : Ctx;
  const float* W = (which == 0) ? Wq : (which == 1 ? Wk : Wv);

  auto stage_copy = [&](int buf, int k0) {
    // A block: 64 rows x 32 f32 = 512 16B chunks, 2 per thread
#pragma unroll
    for (int q = 0; q < 2; ++q) {
      const int c = tid + q * 256;
      const int row = c >> 3, j = (c & 7) * 4;
      cp16_g2s(&lds_a[buf][row][j], A + (size_t)(mblk + row) * kHidden + k0 + j);
    }
    // B block: 128 rows x 32 f32 = 1024 16B chunks, 4 per thread
#pragma unroll
    for (int q = 0; q < 4; ++q) {
      const int c = tid + q * 256;
      const int row = c >> 3, j = (c & 7) * 4;
      cp16_g2s(&lds_b[buf][row][j], W + (size_t)(nblk + row) * kHidden + k0 + j);
    }
  };

  f32x8 acc[2][2] = {};
  constexpr int kSteps = kHidden / 32;  // 32 stages

  stage_copy(0, 0);
  for (int s = 0; s < kSteps; ++s) {
    const int buf = s & 1;
    if (s + 1 < kSteps) {
      stage_copy(buf ^ 1, (s + 1) * 32);
      wait_async<6>();  // 6 newer copies outstanding; stage s complete
    } else {
      wait_async<0>();
    }
    __syncthreads();

    const float* la0 = &lds_a[buf][wm * 32 + l][0];
    const float* la1 = &lds_a[buf][wm * 32 + 16 + l][0];
    const float* lb0 = &lds_b[buf][wn * 32 + l][0];
    const float* lb1 = &lds_b[buf][wn * 32 + 16 + l][0];
    // A 16x32: lane half lo holds K {0..7,16..23}, hi holds {8..15,24..31}
    bf16x16 a0 = cvt2x8(la0 + 8 * hi, la0 + 16 + 8 * hi);
    bf16x16 a1 = cvt2x8(la1 + 8 * hi, la1 + 16 + 8 * hi);
    // B 32x16: lane half lo holds K {0..7,16..23}, hi holds {8..15,24..31} of column n
    bf16x16 b0 = cvt2x8(lb0 + 8 * hi, lb0 + 16 + 8 * hi);
    bf16x16 b1 = cvt2x8(lb1 + 8 * hi, lb1 + 16 + 8 * hi);
    acc[0][0] = wmma_bf16f32(a0, b0, acc[0][0]);
    acc[0][1] = wmma_bf16f32(a0, b1, acc[0][1]);
    acc[1][0] = wmma_bf16f32(a1, b0, acc[1][0]);
    acc[1][1] = wmma_bf16f32(a1, b1, acc[1][1]);

    __syncthreads();  // all waves done reading before buf is overwritten
  }

  // C layout: VGPR i -> row base + i + 8*hi, col = tile base + l.
  // A 32-col wave tile never crosses a 64-wide head boundary.
  const int m0 = mblk + wm * 32;
  const int n0 = nblk + wn * 32;
  if (which == 2) {
#pragma unroll
    for (int nh = 0; nh < 2; ++nh) {
      const int n = n0 + 16 * nh + l;
      const int h = n >> 6, d = n & 63;
#pragma unroll
      for (int mh = 0; mh < 2; ++mh) {
#pragma unroll
        for (int i = 0; i < 8; ++i) {
          const int m = m0 + 16 * mh + i + 8 * hi;
          const int bb = m >> 11, t = m & (kSeq - 1);
          vT_ws[((size_t)(bb * kHeads + h) * kHdim + d) * kSeq + t] =
              (__bf16)acc[mh][nh][i];
        }
      }
    }
  } else {
    __bf16* dst = (which == 0) ? q_ws : k_ws;
    const float sc = (which == 0) ? kScale : 1.0f;  // fold 1/sqrt(d) into Q
#pragma unroll
    for (int nh = 0; nh < 2; ++nh) {
      const int n = n0 + 16 * nh + l;
      const int h = n >> 6, d = n & 63;
#pragma unroll
      for (int mh = 0; mh < 2; ++mh) {
#pragma unroll
        for (int i = 0; i < 8; ++i) {
          const int m = m0 + 16 * mh + i + 8 * hi;
          const int bb = m >> 11, t = m & (kSeq - 1);
          dst[((size_t)(bb * kHeads + h) * kSeq + t) * kHdim + d] =
              (__bf16)(acc[mh][nh][i] * sc);
        }
      }
    }
  }
}

// ---------------------------------------------------------------------------
// Kernel 2: causal flash attention, one wave per 16-row q tile, 32 kv/chunk.
// S = Q K^T via 4 WMMA (2 col-tiles x 2 d-chunks); one online-softmax update
// per 32 kv (S tiles combined elementwise before the shuffle reduction);
// P (16x32, full K - no padding) transposed through per-wave LDS into
// A-fragment layout (b128 DS reads); O += P V via 4 WMMA (d chunks of 16).
// ---------------------------------------------------------------------------
__global__ __launch_bounds__(256) void attn_kernel(
    const __bf16* __restrict__ q_ws, const __bf16* __restrict__ k_ws,
    const __bf16* __restrict__ vT_ws, __bf16* __restrict__ y_ws) {
  __shared__ __align__(16) __bf16 p_lds[8][16][32];  // per-wave P tile

  const int wave = threadIdx.x >> 5;
  const int lane = threadIdx.x & 31;
  const int l = lane & 15, hi = lane >> 4;
  const int qt = blockIdx.x * 8 + wave;  // q tile index 0..127
  const int bh = blockIdx.y;

  const __bf16* Q  = q_ws  + (size_t)bh * kSeq * kHdim;
  const __bf16* K  = k_ws  + (size_t)bh * kSeq * kHdim;
  const __bf16* VT = vT_ws + (size_t)bh * kHdim * kSeq;

  // Q A-fragments for d=[0,32) and [32,64)  (Q already scaled by 1/sqrt(d))
  bf16x16 qa[2];
  {
    const __bf16* qr = Q + (size_t)(qt * 16 + l) * kHdim;
#pragma unroll
    for (int f = 0; f < 2; ++f)
      qa[f] = load2x8(qr + f * 32 + 8 * hi, qr + f * 32 + 16 + 8 * hi);
  }

  f32x8 o[4] = {};
  float rm[8], rl[8];
#pragma unroll
  for (int i = 0; i < 8; ++i) { rm[i] = -__builtin_inff(); rl[i] = 0.0f; }

  const int nc = (qt >> 1) + 1;  // number of 32-wide kv chunks
  for (int c = 0; c < nc; ++c) {
    const int kv0 = c * 32;
    const bool diag = (c == nc - 1);

    if (c + 1 < nc) {  // prefetch next chunk's K/V rows
      __builtin_prefetch(K + (size_t)(kv0 + 32 + l) * kHdim, 0, 0);
      __builtin_prefetch(VT + (size_t)l * kSeq + kv0 + 32, 0, 0);
    }

    // ---- S (16x32) = Q K^T : two 16x16 C tiles ----
    f32x8 s[2] = {};
#pragma unroll
    for (int f2 = 0; f2 < 2; ++f2) {    // kv column tile
#pragma unroll
      for (int f = 0; f < 2; ++f) {     // d chunk of 32
        const __bf16* kr =
            K + (size_t)(kv0 + 16 * f2 + l) * kHdim + f * 32 + 8 * hi;
        s[f2] = wmma_bf16f32(qa[f], load2x8(kr, kr + 16), s[f2]);
      }
    }

    // ---- causal mask (only last chunk touches the diagonal) ----
    float p0[8], p1[8];
#pragma unroll
    for (int i = 0; i < 8; ++i) {
      float a0 = s[0][i], a1 = s[1][i];
      if (diag) {
        const int row = qt * 16 + i + 8 * hi;   // global row
        if (kv0 + l > row)      a0 = -__builtin_inff();
        if (kv0 + 16 + l > row) a1 = -__builtin_inff();
      }
      p0[i] = a0;
      p1[i] = a1;
    }

    // ---- online softmax: one update per 32 kv ----
#pragma unroll
    for (int i = 0; i < 8; ++i) {
      float mx   = half_max(fmaxf(p0[i], p1[i]));
      float mnew = fmaxf(rm[i], mx);
      float alpha = __expf(rm[i] - mnew);       // exp(-inf)=0 on first chunk
      float e0 = __expf(p0[i] - mnew);
      float e1 = __expf(p1[i] - mnew);
      rl[i] = rl[i] * alpha + half_sum(e0 + e1);
      rm[i] = mnew;
      p0[i] = e0;
      p1[i] = e1;
#pragma unroll
      for (int dc = 0; dc < 4; ++dc) o[dc][i] *= alpha;
    }

    // ---- P: C-layout -> A-layout via per-wave LDS transpose ----
#pragma unroll
    for (int i = 0; i < 8; ++i) {
      p_lds[wave][i + 8 * hi][l]      = (__bf16)p0[i];
      p_lds[wave][i + 8 * hi][16 + l] = (__bf16)p1[i];
    }
    asm volatile("s_wait_dscnt 0x0" ::: "memory");  // DS in-order within wave
    // lane lo: K {0..7,16..23}; lane hi: K {8..15,24..31} -> two b128 DS loads
    bf16x16 pa = load2x8(&p_lds[wave][l][8 * hi], &p_lds[wave][l][16 + 8 * hi]);

    // ---- O += P V : B column n = VT row (fixed d, contiguous in t) ----
#pragma unroll
    for (int dc = 0; dc < 4; ++dc) {
      const __bf16* vr = VT + (size_t)(dc * 16 + l) * kSeq + kv0 + 8 * hi;
      bf16x16 vb = load2x8(vr, vr + 16);
      o[dc] = wmma_bf16f32(pa, vb, o[dc]);
    }
  }

  // ---- normalize and scatter to y [b, t, h*64+d] (bf16) ----
  const int b = bh >> 4, h = bh & 15;
#pragma unroll
  for (int dc = 0; dc < 4; ++dc) {
#pragma unroll
    for (int i = 0; i < 8; ++i) {
      const int t = qt * 16 + i + 8 * hi;
      y_ws[((size_t)(b * kSeq + t)) * kHidden + h * kHdim + dc * 16 + l] =
          (__bf16)(o[dc][i] / rl[i]);
    }
  }
}

// ---------------------------------------------------------------------------
// Kernel 3: output projection  out = y @ Wo^T + bo  (fp32 output), 32x32/wave.
// ---------------------------------------------------------------------------
__global__ __launch_bounds__(256) void out_proj_kernel(
    const __bf16* __restrict__ Y, const float* __restrict__ Wo,
    const float* __restrict__ bo, float* __restrict__ out) {
  const int wave = threadIdx.x >> 5;
  const int lane = threadIdx.x & 31;
  const int l = lane & 15, hi = lane >> 4;
  const int m0 = blockIdx.x * 64 + (wave >> 2) * 32;
  const int n0 = blockIdx.y * 128 + (wave & 3) * 32;

  const __bf16* arow0 = Y + (size_t)(m0 + l) * kHidden;
  const __bf16* arow1 = Y + (size_t)(m0 + 16 + l) * kHidden;
  const float*  brow0 = Wo + (size_t)(n0 + l) * kHidden;
  const float*  brow1 = Wo + (size_t)(n0 + 16 + l) * kHidden;

  f32x8 acc[2][2] = {};
  for (int k = 0; k < kHidden; k += 32) {
    __builtin_prefetch(brow0 + k + 256, 0, 0);
    bf16x16 a0 = load2x8(arow0 + k + 8 * hi, arow0 + k + 16 + 8 * hi);
    bf16x16 a1 = load2x8(arow1 + k + 8 * hi, arow1 + k + 16 + 8 * hi);
    bf16x16 b0 = cvt2x8(brow0 + k + 8 * hi, brow0 + k + 16 + 8 * hi);
    bf16x16 b1 = cvt2x8(brow1 + k + 8 * hi, brow1 + k + 16 + 8 * hi);
    acc[0][0] = wmma_bf16f32(a0, b0, acc[0][0]);
    acc[0][1] = wmma_bf16f32(a0, b1, acc[0][1]);
    acc[1][0] = wmma_bf16f32(a1, b0, acc[1][0]);
    acc[1][1] = wmma_bf16f32(a1, b1, acc[1][1]);
  }

#pragma unroll
  for (int nh = 0; nh < 2; ++nh) {
    const int n = n0 + 16 * nh + l;
    const float bias = bo[n];
#pragma unroll
    for (int mh = 0; mh < 2; ++mh) {
#pragma unroll
      for (int i = 0; i < 8; ++i) {
        const int m = m0 + 16 * mh + i + 8 * hi;
        out[(size_t)m * kHidden + n] = acc[mh][nh][i] + bias;
      }
    }
  }
}

// ---------------------------------------------------------------------------
extern "C" void kernel_launch(void* const* d_in, const int* in_sizes, int n_in,
                              void* d_out, int out_size, void* d_ws, size_t ws_size,
                              hipStream_t stream) {
  const float* x   = (const float*)d_in[0];
  const float* ctx = (const float*)d_in[1];
  const float* Wq  = (const float*)d_in[2];
  const float* Wk  = (const float*)d_in[3];
  const float* Wv  = (const float*)d_in[4];
  const float* Wo  = (const float*)d_in[5];
  const float* bo  = (const float*)d_in[6];
  float* out = (float*)d_out;

  const size_t per_buf = (size_t)kBatch * kHeads * kSeq * kHdim;  // 4Mi elems
  __bf16* q_ws  = (__bf16*)d_ws;
  __bf16* k_ws  = q_ws + per_buf;
  __bf16* vT_ws = k_ws + per_buf;
  __bf16* y_ws  = vT_ws + per_buf;   // [MTOT][HIDDEN] bf16

  dim3 gproj(kMtot / 64, kHidden / 128, 3);
  qkv_proj_kernel<<<gproj, 256, 0, stream>>>(x, ctx, Wq, Wk, Wv,
                                             q_ws, k_ws, vT_ws);

  dim3 gattn(kSeq / 16 / 8, kBatch * kHeads);
  attn_kernel<<<gattn, 256, 0, stream>>>(q_ws, k_ws, vT_ws, y_ws);

  dim3 gout(kMtot / 64, kHidden / 128);
  out_proj_kernel<<<gout, 256, 0, stream>>>(y_ws, Wo, bo, out);
}
